// DTIConvGraph3_IGN_basic_45930380264275
// MI455X (gfx1250) — matrix operand order repaired and checked by reading, not compiled, not run on hardware
//
#include <hip/hip_runtime.h>
#include <hip/hip_bf16.h>

typedef _Float16 half_t;
typedef __attribute__((ext_vector_type(16))) _Float16 v16h;
typedef __attribute__((ext_vector_type(8)))  _Float16 v8h;
typedef __attribute__((ext_vector_type(4)))  _Float16 v4h;
typedef __attribute__((ext_vector_type(8)))  float    v8f;

#define EPB 128          // edges per block
#define LDX 264          // 256 + 8 halves pad  (row stride 528B -> 4-bank rotate)
#define LDW 264
#define LDA 136          // 128 + 8 halves pad  (row stride 272B)

// shared-memory carve (in half_t units)
#define SW1_OFF 0                     // 128*264 = 33792
#define SW2_OFF 33792                 // 128*136 = 17408
#define SW3_OFF 51200                 // 17408
#define SX_OFF  68608                 // 128*264 = 33792
#define SACT_OFF 102400               // 17408
#define SBIAS_OFF 119808              // 384 floats after this (byte off 239616)
#define SMEM_BYTES (239616 + 384*4)   // 241152 B  (< 320KB WGP LDS)

__global__ void cvt_weights_f16(const float* __restrict__ W1,
                                const float* __restrict__ W2,
                                const float* __restrict__ W3,
                                half_t* __restrict__ ws) {
    int i = blockIdx.x * 256 + threadIdx.x;           // 0 .. 65535
    if (i < 32768)       ws[i] = (half_t)W1[i];
    else if (i < 49152)  ws[i] = (half_t)W2[i - 32768];
    else                 ws[i] = (half_t)W3[i - 49152];
}

__device__ __forceinline__ v16h frag16(const half_t* p, int gap) {
    v8h lo = *(const v8h*)p;
    v8h hi = *(const v8h*)(p + gap);
    return __builtin_shufflevector(lo, hi, 0,1,2,3,4,5,6,7,8,9,10,11,12,13,14,15);
}

__global__ __launch_bounds__(256) void edge_mlp_wmma(
    const float* __restrict__ e, const float* __restrict__ h,
    const int* __restrict__ src, const int* __restrict__ dst,
    const half_t* __restrict__ w1h, const half_t* __restrict__ w2h,
    const half_t* __restrict__ w3h,
    const float* __restrict__ b1, const float* __restrict__ b2,
    const float* __restrict__ b3,
    float* __restrict__ out, long E) {

    extern __shared__ __align__(16) char smem_raw[];
    half_t* sm   = (half_t*)smem_raw;
    half_t* sW1  = sm + SW1_OFF;
    half_t* sW2  = sm + SW2_OFF;
    half_t* sW3  = sm + SW3_OFF;
    half_t* sX   = sm + SX_OFF;
    half_t* sAct = sm + SACT_OFF;
    float*  sB   = (float*)(sm + SBIAS_OFF);

    const int tid = threadIdx.x;
    const long blockBase = (long)blockIdx.x * EPB;

    // ---- stage f16 weights into padded LDS tiles ----
    for (int c = tid; c < 4096; c += 256) {           // W1: 128x256 in 8-half chunks
        int row = c >> 5, col = (c & 31) << 3;
        *(v8h*)(sW1 + row * LDW + col) = *(const v8h*)(w1h + (row << 8) + col);
    }
    for (int c = tid; c < 2048; c += 256) {           // W2: 128x128
        int row = c >> 4, col = (c & 15) << 3;
        *(v8h*)(sW2 + row * LDA + col) = *(const v8h*)(w2h + (row << 7) + col);
    }
    for (int c = tid; c < 2048; c += 256) {           // W3: 128x128
        int row = c >> 4, col = (c & 15) << 3;
        *(v8h*)(sW3 + row * LDA + col) = *(const v8h*)(w3h + (row << 7) + col);
    }
    for (int i = tid; i < 384; i += 256)
        sB[i] = (i < 128) ? b1[i] : (i < 256) ? b2[i - 128] : b3[i - 256];

    // ---- stage X = [e | h[src]+h[dst]] as f16, 2 threads per edge ----
    {
        int m = tid >> 1, hf = tid & 1;
        long ge = blockBase + m;
        long gc = (ge < E) ? ge : (E - 1);
        int si = src[gc], di = dst[gc];
        const float4* ep = (const float4*)(e + gc * 128 + hf * 64);
        const float4* hs = (const float4*)(h + (long)si * 128 + hf * 64);
        const float4* hd = (const float4*)(h + (long)di * 128 + hf * 64);
        half_t* xe = sX + m * LDX + hf * 64;
        half_t* xh = sX + m * LDX + 128 + hf * 64;
#pragma unroll
        for (int i = 0; i < 16; ++i) {
            float4 v = ep[i];
            v4h hv = {(half_t)v.x, (half_t)v.y, (half_t)v.z, (half_t)v.w};
            *(v4h*)(xe + i * 4) = hv;
        }
#pragma unroll
        for (int i = 0; i < 16; ++i) {
            float4 a = hs[i], b = hd[i];
            v4h hv = {(half_t)(a.x + b.x), (half_t)(a.y + b.y),
                      (half_t)(a.z + b.z), (half_t)(a.w + b.w)};
            *(v4h*)(xh + i * 4) = hv;
        }
    }
    __syncthreads();

    const int lane = tid & 31;
    const int wv   = tid >> 5;
    const int m0   = wv << 4;        // 16-edge M tile per wave
    const int lr   = lane & 15;
    const int hi   = lane >> 4;      // lane half selects K subset

    // ================= layer 1 : K=256 =================
    v16h a1[8];
#pragma unroll
    for (int ks = 0; ks < 8; ++ks)   // A layout: lane<16 K{0..7,16..23}, lane>=16 +8
        a1[ks] = frag16(sX + (m0 + lr) * LDX + ks * 32 + hi * 8, 16);

#pragma unroll
    for (int nt = 0; nt < 8; ++nt) {
        v8f acc = {0.f, 0.f, 0.f, 0.f, 0.f, 0.f, 0.f, 0.f};
#pragma unroll
        for (int ks = 0; ks < 8; ++ks) {
            // B layout: lane<16 K{k0..k0+7,k0+16..k0+23}, lane>=16 +8
            v16h b = frag16(sW1 + (nt * 16 + lr) * LDW + ks * 32 + hi * 8, 16);
            acc = __builtin_amdgcn_wmma_f32_16x16x32_f16(false, a1[ks], false, b,
                                                         (short)0, acc, false, false);
        }
        float bia = sB[nt * 16 + lr];
        int mb = m0 + hi * 8;
#pragma unroll
        for (int r = 0; r < 8; ++r) {
            float v = acc[r] + bia;
            v = fmaxf(v, 0.f) + 0.01f * fminf(v, 0.f);
            sAct[(mb + r) * LDA + nt * 16 + lr] = (half_t)v;
        }
    }

    // ================= layer 2 : K=128 =================
    v16h a2[4];
#pragma unroll
    for (int ks = 0; ks < 4; ++ks)
        a2[ks] = frag16(sAct + (m0 + lr) * LDA + ks * 32 + hi * 8, 16);

#pragma unroll
    for (int nt = 0; nt < 8; ++nt) {
        v8f acc = {0.f, 0.f, 0.f, 0.f, 0.f, 0.f, 0.f, 0.f};
#pragma unroll
        for (int ks = 0; ks < 4; ++ks) {
            v16h b = frag16(sW2 + (nt * 16 + lr) * LDA + ks * 32 + hi * 8, 16);
            acc = __builtin_amdgcn_wmma_f32_16x16x32_f16(false, a2[ks], false, b,
                                                         (short)0, acc, false, false);
        }
        float bia = sB[128 + nt * 16 + lr];
        int mb = m0 + hi * 8;
#pragma unroll
        for (int r = 0; r < 8; ++r) {
            float v = acc[r] + bia;
            v = fmaxf(v, 0.f) + 0.01f * fminf(v, 0.f);
            sAct[(mb + r) * LDA + nt * 16 + lr] = (half_t)v;
        }
    }

    // ================= layer 3 : K=128, store f32 =================
    v16h a3[4];
#pragma unroll
    for (int ks = 0; ks < 4; ++ks)
        a3[ks] = frag16(sAct + (m0 + lr) * LDA + ks * 32 + hi * 8, 16);

#pragma unroll
    for (int nt = 0; nt < 8; ++nt) {
        v8f acc = {0.f, 0.f, 0.f, 0.f, 0.f, 0.f, 0.f, 0.f};
#pragma unroll
        for (int ks = 0; ks < 4; ++ks) {
            v16h b = frag16(sW3 + (nt * 16 + lr) * LDA + ks * 32 + hi * 8, 16);
            acc = __builtin_amdgcn_wmma_f32_16x16x32_f16(false, a3[ks], false, b,
                                                         (short)0, acc, false, false);
        }
        float bia = sB[256 + nt * 16 + lr];
        int mb = m0 + hi * 8;
#pragma unroll
        for (int r = 0; r < 8; ++r) {
            float v = acc[r] + bia;
            v = fmaxf(v, 0.f) + 0.01f * fminf(v, 0.f);
            long ge = blockBase + mb + r;
            if (ge < E)
                out[ge * 128 + nt * 16 + lr] = v;
        }
    }
}

extern "C" void kernel_launch(void* const* d_in, const int* in_sizes, int n_in,
                              void* d_out, int out_size, void* d_ws, size_t ws_size,
                              hipStream_t stream) {
    (void)n_in; (void)out_size; (void)ws_size;
    const float* e  = (const float*)d_in[0];
    const float* h  = (const float*)d_in[1];
    const int* src  = (const int*)d_in[2];
    const int* dst  = (const int*)d_in[3];
    const float* W1 = (const float*)d_in[4];
    const float* b1 = (const float*)d_in[5];
    const float* W2 = (const float*)d_in[6];
    const float* b2 = (const float*)d_in[7];
    const float* W3 = (const float*)d_in[8];
    const float* b3 = (const float*)d_in[9];
    float* out = (float*)d_out;
    half_t* wsh = (half_t*)d_ws;   // 65536 halves = 128KB scratch for f16 weights

    long E = in_sizes[2];

    cvt_weights_f16<<<256, 256, 0, stream>>>(W1, W2, W3, wsh);

    int nblk = (int)((E + EPB - 1) / EPB);
    edge_mlp_wmma<<<nblk, 256, SMEM_BYTES, stream>>>(
        e, h, src, dst, wsh, wsh + 32768, wsh + 49152, b1, b2, b3, out, E);
}
